// Block_performer_78520592105821
// MI455X (gfx1250) — matrix operand order repaired and checked by reading, not compiled, not run on hardware
//
#include <hip/hip_runtime.h>

typedef __attribute__((ext_vector_type(16))) __bf16 v16bf;
typedef __attribute__((ext_vector_type(8)))  float  v8f;

#define BATCH   4
#define TSEQ    4096
#define DMODEL  1024
#define MFEAT   512
#define ROWS    (BATCH * TSEQ)
#define LDT     40                 // LDS tile leading dim (elems): 80B, 16B aligned
#define INV_SQRT_M 0.044194173824159216f
#define EPS_D   1e-8f

// ---------- bf16 helpers ----------
// Native packed convert: 2 floats -> 2 bf16 per instruction (vs 3 VALU ops/elem manual RNE)
__device__ __forceinline__ unsigned cvt_pk_bf16(float lo, float hi) {
  unsigned r;
  asm("v_cvt_pk_bf16_f32 %0, %1, %2" : "=v"(r) : "v"(lo), "v"(hi));
  return r;
}
__device__ __forceinline__ unsigned short bf_from_f32(float f) {
  return (unsigned short)cvt_pk_bf16(f, f);
}
__device__ __forceinline__ float f32_from_bf(unsigned short h) {
  return __uint_as_float(((unsigned)h) << 16);
}

union Frag16 { v16bf v; uint4 q[2]; unsigned short h[16]; };

// ---------- WMMA fragment loads from LDS ----------
// A (16x32 bf16), tile stored [row][K], ld = LDT.
// Lanes 0-15: row = lane, K = {0..7, 16..23}; lanes 16-31: row = lane-16, K = {8..15, 24..31}.
__device__ __forceinline__ v16bf load_a_frag(const unsigned short* tile) {
  const int lane = threadIdx.x & 31;
  const unsigned short* p = tile + (lane & 15) * LDT + ((lane & 16) ? 8 : 0);
  Frag16 f;
  f.q[0] = *(const uint4*)(p);        // K base .. base+7
  f.q[1] = *(const uint4*)(p + 16);   // K base+16 .. base+23
  return f.v;
}
// B (32x16 bf16), tile stored [col][K], ld = LDT.
// Lane c = lane&15 holds column c; lanes 0-15 K = {0..7, 16..23}, lanes 16-31 K = {8..15, 24..31}.
__device__ __forceinline__ v16bf load_b_frag(const unsigned short* tile) {
  const int lane = threadIdx.x & 31;
  const unsigned short* p = tile + (lane & 15) * LDT + ((lane & 16) ? 8 : 0);
  Frag16 f;
  f.q[0] = *(const uint4*)(p);
  f.q[1] = *(const uint4*)(p + 16);
  return f.v;
}

// ---------- shared wave-level 128x128x32 step: 8 waves, each 64x32 = 4x2 WMMAs ----------
__device__ __forceinline__ void wave_mma(const unsigned short* As, const unsigned short* Bs,
                                         v8f acc[4][2]) {
  const int wave = threadIdx.x >> 5;
  const int wr = (wave & 1) * 64;
  const int wc = (wave >> 1) * 32;
  v16bf a[4], b[2];
#pragma unroll
  for (int i = 0; i < 4; ++i) a[i] = load_a_frag(As + (wr + i * 16) * LDT);
#pragma unroll
  for (int j = 0; j < 2; ++j) b[j] = load_b_frag(Bs + (wc + j * 16) * LDT);
#pragma unroll
  for (int i = 0; i < 4; ++i)
#pragma unroll
    for (int j = 0; j < 2; ++j)
      acc[i][j] = __builtin_amdgcn_wmma_f32_16x16x32_bf16(
          false, a[i], false, b[j], (short)0, acc[i][j], false, false);
}

// ---------- LDS staging helpers (16 contiguous elems per thread) ----------
__device__ __forceinline__ void stage16_from_f32(unsigned short* dst, const float* src) {
  union { float4 v4[4]; float f[16]; } uf;
  const float4* s = (const float4*)src;
  uf.v4[0] = s[0]; uf.v4[1] = s[1]; uf.v4[2] = s[2]; uf.v4[3] = s[3];
  union { unsigned u[8]; uint4 q[2]; } up;
#pragma unroll
  for (int i = 0; i < 8; ++i) up.u[i] = cvt_pk_bf16(uf.f[2 * i], uf.f[2 * i + 1]);
  ((uint4*)dst)[0] = up.q[0];
  ((uint4*)dst)[1] = up.q[1];
}

// Async global->LDS copy of 32 bytes (2 x b128), bypassing VGPRs. ASYNCcnt tracked.
// Both LDS and global sides add INST_OFFSET (ISA 10: async load pseudocode), so one
// base pair + offset:16 covers the second 16B.
typedef __attribute__((address_space(3))) unsigned short lds_ushort_t;
__device__ __forceinline__ void async_copy16_bf(unsigned short* dst, const unsigned short* src) {
  unsigned lds_off = (unsigned)(unsigned long long)(lds_ushort_t*)dst;
  asm volatile("global_load_async_to_lds_b128 %0, %1, off\n\t"
               "global_load_async_to_lds_b128 %0, %1, off offset:16"
               :: "v"(lds_off), "v"(src) : "memory");
}
__device__ __forceinline__ void wait_async_done() {
  asm volatile("s_wait_asynccnt 0x0" ::: "memory");
}

// ---------- 1) row half-norms: xd[r] = 0.5 * sum(x[r,:]^2) ----------
__global__ void __launch_bounds__(256) halfnorm_kernel(const float* __restrict__ x,
                                                       float* __restrict__ xd) {
  const int row  = blockIdx.x * 8 + (threadIdx.x >> 5);
  const int lane = threadIdx.x & 31;
  const float* p = x + (size_t)row * DMODEL;
  float s = 0.f;
  for (int i = lane; i < DMODEL; i += 32) { float v = p[i]; s += v * v; }
#pragma unroll
  for (int off = 16; off; off >>= 1) s += __shfl_xor(s, off, 32);
  if (lane == 0) xd[row] = 0.5f * s;
}

// ---------- 2) P = exp(X @ W^T - xd) * 1/sqrt(m), bf16 out. X:[ROWS,1024], W:[512,1024] ----------
__global__ void __launch_bounds__(256) prm_gemm_kernel(const float* __restrict__ X,
                                                       const float* __restrict__ W,
                                                       const float* __restrict__ xd,
                                                       unsigned short* __restrict__ P) {
  __shared__ __align__(16) unsigned short As[128 * LDT];
  __shared__ __align__(16) unsigned short Bs[128 * LDT];
  const int row0 = blockIdx.y * 128;
  const int col0 = blockIdx.x * 128;
  const int tid  = threadIdx.x;
  const int ar = tid >> 1, ac = (tid & 1) * 16;    // 128 rows x 32 K staging map

  const v8f zero = {0.f, 0.f, 0.f, 0.f, 0.f, 0.f, 0.f, 0.f};
  v8f acc[4][2];
#pragma unroll
  for (int i = 0; i < 4; ++i) { acc[i][0] = zero; acc[i][1] = zero; }

  for (int k0 = 0; k0 < DMODEL; k0 += 32) {
    stage16_from_f32(&As[ar * LDT + ac], X + (size_t)(row0 + ar) * DMODEL + k0 + ac);
    stage16_from_f32(&Bs[ar * LDT + ac], W + (size_t)(col0 + ar) * DMODEL + k0 + ac);
    if (k0 + 32 < DMODEL)
      __builtin_prefetch(X + (size_t)(row0 + ar) * DMODEL + k0 + 32 + ac);
    __syncthreads();
    wave_mma(As, Bs, acc);
    __syncthreads();
  }
  const int wave = tid >> 5, lane = tid & 31;
  const int wr = (wave & 1) * 64, wc = (wave >> 1) * 32;
  const int rb = ((lane >> 4) & 1) * 8, cl = lane & 15;
#pragma unroll
  for (int i = 0; i < 4; ++i)
#pragma unroll
    for (int e = 0; e < 8; ++e) {
      const int gr = row0 + wr + i * 16 + rb + e;
      const float xv = xd[gr];
#pragma unroll
      for (int j = 0; j < 2; ++j) {
        const int gc = col0 + wc + j * 16 + cl;
        const float p = expf(acc[i][j][e] - xv) * INV_SQRT_M;
        P[(size_t)gr * MFEAT + gc] = bf_from_f32(p);
      }
    }
}

// ---------- 3) deterministic ksum[b,m] = sum_t kp[b,t,m] (two-stage, no atomics) ----------
__global__ void __launch_bounds__(256) ksum_partial_kernel(const unsigned short* __restrict__ KP,
                                                           float* __restrict__ part) {
  const int m = blockIdx.x * 256 + threadIdx.x;   // [0,512)
  const int s = blockIdx.y;                       // 16 chunks of 256 t's
  const int b = blockIdx.z;
  const unsigned short* base = KP + ((size_t)b * TSEQ + s * 256) * MFEAT + m;
  float acc = 0.f;
  for (int t = 0; t < 256; ++t) acc += f32_from_bf(base[(size_t)t * MFEAT]);
  part[((size_t)b * 16 + s) * MFEAT + m] = acc;
}
__global__ void __launch_bounds__(256) ksum_final_kernel(const float* __restrict__ part,
                                                         float* __restrict__ ksum) {
  const int m = blockIdx.x * 256 + threadIdx.x;
  const int b = blockIdx.y;
  float acc = 0.f;
#pragma unroll
  for (int s = 0; s < 16; ++s) acc += part[((size_t)b * 16 + s) * MFEAT + m];
  ksum[b * MFEAT + m] = acc;
}

// ---------- 4) D[b,t] = dot(qp[b,t,:], ksum[b,:]) ----------
__global__ void __launch_bounds__(256) dvec_kernel(const unsigned short* __restrict__ QP,
                                                   const float* __restrict__ ksum,
                                                   float* __restrict__ Dv) {
  const int row  = blockIdx.x * 8 + (threadIdx.x >> 5);
  const int b    = row >> 12;                     // T = 4096
  const int lane = threadIdx.x & 31;
  const unsigned short* q = QP + (size_t)row * MFEAT;
  const float* ks = ksum + b * MFEAT;
  float s = 0.f;
  for (int i = lane; i < MFEAT; i += 32) s += f32_from_bf(q[i]) * ks[i];
#pragma unroll
  for (int off = 16; off; off >>= 1) s += __shfl_xor(s, off, 32);
  if (lane == 0) Dv[row] = s;
}

// ---------- 5) Z[b,m,n] = sum_t kp[b,t,m] * v[b,t,n]  ([512,4096]x[4096,1024]) ----------
__global__ void __launch_bounds__(256) kptv_gemm_kernel(const unsigned short* __restrict__ KP,
                                                        const float* __restrict__ V,
                                                        unsigned short* __restrict__ Z) {
  __shared__ __align__(16) unsigned short As[128 * LDT];
  __shared__ __align__(16) unsigned short Bs[128 * LDT];
  const int b  = blockIdx.z;
  const int m0 = blockIdx.y * 128;
  const int n0 = blockIdx.x * 128;
  const int tid = threadIdx.x;
  const int tt = tid >> 3, cs = (tid & 7) * 16;   // 32 t's x 128 outdim staging (transpose)

  const v8f zero = {0.f, 0.f, 0.f, 0.f, 0.f, 0.f, 0.f, 0.f};
  v8f acc[4][2];
#pragma unroll
  for (int i = 0; i < 4; ++i) { acc[i][0] = zero; acc[i][1] = zero; }

  for (int t0 = 0; t0 < TSEQ; t0 += 32) {
    const unsigned short* asrc = KP + ((size_t)b * TSEQ + t0 + tt) * MFEAT + m0 + cs;
    const float* bsrc = V + ((size_t)b * TSEQ + t0 + tt) * DMODEL + n0 + cs;
#pragma unroll
    for (int i = 0; i < 16; ++i) {
      As[(cs + i) * LDT + tt] = asrc[i];
      Bs[(cs + i) * LDT + tt] = bf_from_f32(bsrc[i]);
    }
    if (t0 + 32 < TSEQ) __builtin_prefetch(bsrc + 32 * DMODEL);
    __syncthreads();
    wave_mma(As, Bs, acc);
    __syncthreads();
  }
  const int wave = tid >> 5, lane = tid & 31;
  const int wr = (wave & 1) * 64, wc = (wave >> 1) * 32;
  const int rb = ((lane >> 4) & 1) * 8, cl = lane & 15;
#pragma unroll
  for (int i = 0; i < 4; ++i)
#pragma unroll
    for (int j = 0; j < 2; ++j)
#pragma unroll
      for (int e = 0; e < 8; ++e) {
        const int gm = m0 + wr + i * 16 + rb + e;
        const int gn = n0 + wc + j * 16 + cl;
        Z[((size_t)b * MFEAT + gm) * DMODEL + gn] = bf_from_f32(acc[i][j][e]);
      }
}

// ---------- 6) Y[b,t,n] = (qp[b,t,:] @ Z[b,:,n]) / (D[b,t]+eps), bf16 out ----------
__global__ void __launch_bounds__(256) y_gemm_kernel(const unsigned short* __restrict__ QP,
                                                     const unsigned short* __restrict__ Z,
                                                     const float* __restrict__ Dv,
                                                     unsigned short* __restrict__ Y) {
  __shared__ __align__(16) unsigned short As[128 * LDT];
  __shared__ __align__(16) unsigned short Bs[128 * LDT];
  const int b   = blockIdx.z;
  const int t0r = blockIdx.y * 128;
  const int n0  = blockIdx.x * 128;
  const int tid = threadIdx.x;
  const int ar = tid >> 1, ac = (tid & 1) * 16;   // A staging (row-major async copy)
  const int bk = tid >> 3, bn = (tid & 7) * 16;   // B staging (transpose from [m][n])

  const v8f zero = {0.f, 0.f, 0.f, 0.f, 0.f, 0.f, 0.f, 0.f};
  v8f acc[4][2];
#pragma unroll
  for (int i = 0; i < 4; ++i) { acc[i][0] = zero; acc[i][1] = zero; }

  for (int k0 = 0; k0 < MFEAT; k0 += 32) {
    // A tile: pure bf16 copy -> async global->LDS (no VGPR round trip)
    async_copy16_bf(&As[ar * LDT + ac],
                    QP + ((size_t)b * TSEQ + t0r + ar) * MFEAT + k0 + ac);
    const unsigned short* bsrc = Z + ((size_t)b * MFEAT + k0 + bk) * DMODEL + n0 + bn;
#pragma unroll
    for (int i = 0; i < 16; ++i) Bs[(bn + i) * LDT + bk] = bsrc[i];
    wait_async_done();
    __syncthreads();
    wave_mma(As, Bs, acc);
    __syncthreads();
  }
  const int wave = tid >> 5, lane = tid & 31;
  const int wr = (wave & 1) * 64, wc = (wave >> 1) * 32;
  const int rb = ((lane >> 4) & 1) * 8, cl = lane & 15;
#pragma unroll
  for (int i = 0; i < 4; ++i)
#pragma unroll
    for (int e = 0; e < 8; ++e) {
      const int gt = t0r + wr + i * 16 + rb + e;
      const float inv = 1.0f / (Dv[b * TSEQ + gt] + EPS_D);
#pragma unroll
      for (int j = 0; j < 2; ++j) {
        const int gn = n0 + wc + j * 16 + cl;
        Y[((size_t)b * TSEQ + gt) * DMODEL + gn] = bf_from_f32(acc[i][j][e] * inv);
      }
    }
}

// ---------- 7) out = Y @ proj_w^T, f32 out. Y:[ROWS,1024] bf16, PW:[1024,1024] f32 ----------
__global__ void __launch_bounds__(256) proj_gemm_kernel(const unsigned short* __restrict__ Y,
                                                        const float* __restrict__ PW,
                                                        float* __restrict__ out) {
  __shared__ __align__(16) unsigned short As[128 * LDT];
  __shared__ __align__(16) unsigned short Bs[128 * LDT];
  const int row0 = blockIdx.y * 128;
  const int n0   = blockIdx.x * 128;
  const int tid  = threadIdx.x;
  const int ar = tid >> 1, ac = (tid & 1) * 16;

  const v8f zero = {0.f, 0.f, 0.f, 0.f, 0.f, 0.f, 0.f, 0.f};
  v8f acc[4][2];
#pragma unroll
  for (int i = 0; i < 4; ++i) { acc[i][0] = zero; acc[i][1] = zero; }

  for (int k0 = 0; k0 < DMODEL; k0 += 32) {
    // A tile: pure bf16 copy -> async global->LDS
    async_copy16_bf(&As[ar * LDT + ac], Y + (size_t)(row0 + ar) * DMODEL + k0 + ac);
    stage16_from_f32(&Bs[ar * LDT + ac], PW + (size_t)(n0 + ar) * DMODEL + k0 + ac);
    if (k0 + 32 < DMODEL)
      __builtin_prefetch(PW + (size_t)(n0 + ar) * DMODEL + k0 + 32 + ac);
    wait_async_done();
    __syncthreads();
    wave_mma(As, Bs, acc);
    __syncthreads();
  }
  const int wave = tid >> 5, lane = tid & 31;
  const int wr = (wave & 1) * 64, wc = (wave >> 1) * 32;
  const int rb = ((lane >> 4) & 1) * 8, cl = lane & 15;
#pragma unroll
  for (int i = 0; i < 4; ++i)
#pragma unroll
    for (int j = 0; j < 2; ++j)
#pragma unroll
      for (int e = 0; e < 8; ++e) {
        const int gr = row0 + wr + i * 16 + rb + e;
        const int gc = n0 + wc + j * 16 + cl;
        out[(size_t)gr * DMODEL + gc] = acc[i][j][e];
      }
}

// ---------------------------------------------------------------------------
extern "C" void kernel_launch(void* const* d_in, const int* in_sizes, int n_in,
                              void* d_out, int out_size, void* d_ws, size_t ws_size,
                              hipStream_t stream) {
  (void)in_sizes; (void)n_in; (void)out_size; (void)ws_size;
  const float* q  = (const float*)d_in[0];
  const float* k  = (const float*)d_in[1];
  const float* v  = (const float*)d_in[2];
  const float* w  = (const float*)d_in[3];
  const float* pw = (const float*)d_in[4];
  float* out = (float*)d_out;

  size_t off = 0;
  char* ws = (char*)d_ws;
  auto carve = [&](size_t bytes) -> void* {
    void* p = (void*)(ws + off);
    off += (bytes + 255) & ~(size_t)255;
    return p;
  };
  float* xk   = (float*)carve((size_t)ROWS * sizeof(float));
  float* xq   = (float*)carve((size_t)ROWS * sizeof(float));
  float* part = (float*)carve((size_t)BATCH * 16 * MFEAT * sizeof(float));
  float* ksum = (float*)carve((size_t)BATCH * MFEAT * sizeof(float));
  float* Dv   = (float*)carve((size_t)ROWS * sizeof(float));
  unsigned short* KP = (unsigned short*)carve((size_t)ROWS * MFEAT * 2);
  unsigned short* QP = (unsigned short*)carve((size_t)ROWS * MFEAT * 2);
  unsigned short* Z  = (unsigned short*)carve((size_t)BATCH * MFEAT * DMODEL * 2);
  unsigned short* Y  = (unsigned short*)carve((size_t)ROWS * DMODEL * 2);

  // 1) half squared norms
  halfnorm_kernel<<<ROWS / 8, 256, 0, stream>>>(k, xk);
  halfnorm_kernel<<<ROWS / 8, 256, 0, stream>>>(q, xq);

  // 2) random-feature maps (bf16 WMMA GEMM + exp epilogue)
  prm_gemm_kernel<<<dim3(MFEAT / 128, ROWS / 128), 256, 0, stream>>>(k, w, xk, KP);
  prm_gemm_kernel<<<dim3(MFEAT / 128, ROWS / 128), 256, 0, stream>>>(q, w, xq, QP);

  // 3) deterministic column-sum of kp
  ksum_partial_kernel<<<dim3(MFEAT / 256, 16, BATCH), 256, 0, stream>>>(KP, part);
  ksum_final_kernel<<<dim3(MFEAT / 256, BATCH), 256, 0, stream>>>(part, ksum);

  // 4) denominator D[b,t]
  dvec_kernel<<<ROWS / 8, 256, 0, stream>>>(QP, ksum, Dv);

  // 5) Z = kp^T @ v per batch
  kptv_gemm_kernel<<<dim3(DMODEL / 128, MFEAT / 128, BATCH), 256, 0, stream>>>(KP, v, Z);

  // 6) Y = (qp @ Z) / (D + eps)
  y_gemm_kernel<<<dim3(DMODEL / 128, TSEQ / 128, BATCH), 256, 0, stream>>>(QP, Z, Dv, Y);

  // 7) out = Y @ proj_w^T (f32)
  proj_gemm_kernel<<<dim3(DMODEL / 128, ROWS / 128), 256, 0, stream>>>(Y, pw, out);
}
